// GaussianMixture_64931315581666
// MI455X (gfx1250) — compile-verified, matrix operand order checked by reading, not run on hardware
//
#include <hip/hip_runtime.h>

typedef __attribute__((ext_vector_type(16))) _Float16 v16h;
typedef __attribute__((ext_vector_type(8)))  float    v8f;

#define NPTS 131072
#define KCL  256
#define DIM  32
#define GRPCH 9                      // chunks per copy-group (33 real -> 4 groups of 9)
#define NTILE 4                      // ktiles per kgroup
#define GRPBYTES (NTILE * GRPCH * 1024)   // 36864 B per group

// Workspace layout:
//   [0, 1KB)        : g[K] per-cluster log-coefficient constants
//   [1KB, 1KB+576KB): packed WMMA B fragments, grouped for linear async copy:
//     [kg 0..3][grp 0..3][tile 0..3][chunk c 0..8][1 KB fragment]
//     global chunk d = grp*9 + c ; d in 0..31 = rows of A_k ; d==32 = -2*m_k ;
//     d in 33..35 = pad (never read).
//     fragment lane slot = lane*32B : lane n (n<16): cluster kt*16+n,
//     e-set {0..7,16..23}; lane n+16: same cluster, e-set {8..15,24..31}.

__device__ __forceinline__ size_t frag_off(int kt, int d) {
    const int kg = kt >> 2, t4 = kt & 3;
    const int grp = d / GRPCH, c = d - grp * GRPCH;
    return ((size_t)((kg * 16 + grp * 4 + t4) * GRPCH + c)) << 10;
}

__global__ __launch_bounds__(32) void gmm_prep(
    const float* __restrict__ centers,
    const float* __restrict__ covs_inv_sqrt,
    const float* __restrict__ weights,
    float* __restrict__ gout,
    char* __restrict__ bpack)
{
    const int k = blockIdx.x;
    const int t = threadIdx.x;            // 0..31, one wave

    __shared__ float S[DIM][DIM];
    __shared__ float A[DIM][DIM + 1];
    __shared__ float mv[DIM];

    const float* Sg = covs_inv_sqrt + (size_t)k * DIM * DIM;
    for (int i = t; i < DIM * DIM; i += 32) (&S[0][0])[i] = Sg[i];
    __syncthreads();

    // A = S * S^T  (thread t owns row t)
    for (int f = 0; f < DIM; ++f) {
        float acc = 0.f;
        for (int e = 0; e < DIM; ++e) acc += S[t][e] * S[f][e];
        A[t][f] = acc;
    }
    __syncthreads();

    // Pack WMMA B fragment for chunk d = t
    {
        const int n = k & 15, kt = k >> 4;
        for (int slot = 0; slot < 2; ++slot) {
            _Float16* dst = (_Float16*)(bpack + frag_off(kt, t) + (size_t)(n + slot * 16) * 32);
            for (int j = 0; j < 16; ++j) {
                int e = (j < 8 ? j : j + 8) + slot * 8;
                dst[j] = (_Float16)A[t][e];
            }
        }
    }

    // m = A * c  (thread t owns component t)
    const float* ck = centers + (size_t)k * DIM;
    {
        float acc = 0.f;
        for (int e = 0; e < DIM; ++e) acc += A[t][e] * ck[e];
        mv[t] = acc;
    }
    __syncthreads();

    // chunk-32 fragment: -2*m (folds the cross term into the GEMM)
    if (t < 2) {
        const int n = k & 15, kt = k >> 4;
        _Float16* dst = (_Float16*)(bpack + frag_off(kt, 32) + (size_t)(n + t * 16) * 32);
        for (int j = 0; j < 16; ++j) {
            int e = (j < 8 ? j : j + 8) + t * 8;
            dst[j] = (_Float16)(-2.0f * mv[e]);
        }
    }

    // log|A| via no-pivot Gaussian elimination (A is SPD -> positive pivots)
    for (int p = 0; p < DIM - 1; ++p) {
        __syncthreads();
        if (t > p) {
            float f = A[t][p] / A[p][p];
            for (int c2 = p; c2 < DIM; ++c2) A[t][c2] -= f * A[p][c2];
        }
    }
    __syncthreads();

    if (t == 0) {
        float logdet = 0.f;
        for (int p = 0; p < DIM; ++p) logdet += __logf(fabsf(A[p][p]) + 1e-37f);
        float wsum = 0.f;
        for (int j = 0; j < KCL; ++j) wsum += fabsf(weights[j]);
        float pr = fabsf(weights[k]) / (wsum + 1e-30f);
        float tcac = 0.f;
        for (int e = 0; e < DIM; ++e) tcac += mv[e] * ck[e];
        gout[k] = __logf(pr + 1e-37f) + 0.5f * logdet - 0.5f * tcac;
    }
}

__global__ __launch_bounds__(256) void gmm_ll(
    const float* __restrict__ points,
    const float* __restrict__ gvals,
    const char* __restrict__ bpack,
    const float* __restrict__ thr,
    float* __restrict__ out)
{
    const int tid  = threadIdx.x;
    const int wave = tid >> 5;
    const int lane = tid & 31;
    const int row0 = (blockIdx.x * 8 + wave) * 32;    // 32 points per wave

    __shared__ float xs[8][32 * DIM];                 // 4 KB x-tile per wave
    __shared__ __align__(32) char bstage[2][GRPBYTES];// double-buffered B stage

    // issue one copy-group's async global->LDS transfer (9 issues per wave)
    auto issue = [&](int gi) {
        const unsigned lds0 = (unsigned)(size_t)(void*)&bstage[gi & 1][0];
        const char* gbase = bpack + (size_t)gi * GRPBYTES;
        #pragma unroll
        for (int s = 0; s < 9; ++s) {
            const unsigned off = (unsigned)(s * 4096 + tid * 16);
            const unsigned lds = lds0 + off;
            const unsigned long long ga = (unsigned long long)(gbase + off);
            asm volatile("global_load_async_to_lds_b128 %0, %1, off"
                         :: "v"(lds), "v"(ga) : "memory");
        }
    };

    issue(0);   // prefetch first group while we stage the x-tile

    float* xw = xs[wave];
    {
        const float4* src = (const float4*)(points + (size_t)row0 * DIM);
        float4* dst4 = (float4*)xw;
        for (int i = lane; i < (32 * DIM) / 4; i += 32) dst4[i] = src[i];
    }

    const int rlo   = lane & 15;
    const int upper = lane >> 4;
    const float* xr0 = xw + rlo * DIM;          // M-tile 0 row
    const float* xr1 = xw + (rlo + 16) * DIM;   // M-tile 1 row

    __syncthreads();

    v16h xe0, xe1;
    #pragma unroll
    for (int j = 0; j < 16; ++j) {
        int e = (j < 8 ? j : j + 8) + upper * 8;
        xe0[j] = (_Float16)xr0[e];
        xe1[j] = (_Float16)xr1[e];
    }

    float mrun[2][8], srun[2][8];
    #pragma unroll
    for (int mt = 0; mt < 2; ++mt)
        #pragma unroll
        for (int r = 0; r < 8; ++r) { mrun[mt][r] = -3.0e38f; srun[mt][r] = 0.f; }

    v8f zero = {};
    v8f acc[NTILE][2];
    #pragma unroll
    for (int tt = 0; tt < NTILE; ++tt) { acc[tt][0] = zero; acc[tt][1] = zero; }

    // one chunk: load all 4 B-fragments (8 ds_loads in flight), then 8 WMMAs
    auto do_chunk = [&](const char* bw, int c, v16h af0, v16h af1) {
        v16h bf[NTILE];
        #pragma unroll
        for (int tt = 0; tt < NTILE; ++tt)
            bf[tt] = *(const v16h*)(bw + (size_t)(tt * GRPCH + c) * 1024
                                       + (size_t)lane * 32);
        #pragma unroll
        for (int tt = 0; tt < NTILE; ++tt) {
            acc[tt][0] = __builtin_amdgcn_wmma_f32_16x16x32_f16(
                false, af0, false, bf[tt], (short)0, acc[tt][0], false, false);
            acc[tt][1] = __builtin_amdgcn_wmma_f32_16x16x32_f16(
                false, af1, false, bf[tt], (short)0, acc[tt][1], false, false);
        }
    };

    for (int gi = 0; gi < 16; ++gi) {
        const int kg = gi >> 2, grp = gi & 3;

        if (gi + 1 < 16) {
            issue(gi + 1);                                   // prefetch next group
            asm volatile("s_wait_asynccnt 0x9" ::: "memory"); // current group done
        } else {
            asm volatile("s_wait_asynccnt 0x0" ::: "memory");
        }
        __syncthreads();

        const char* bw = &bstage[gi & 1][0];
        if (grp < 3) {
            #pragma unroll
            for (int c = 0; c < GRPCH; ++c) {
                const int d = grp * GRPCH + c;              // always < 27
                _Float16 x0 = (_Float16)xr0[d];
                _Float16 x1 = (_Float16)xr1[d];
                do_chunk(bw, c, xe0 * x0, xe1 * x1);        // a[m][e]=x[m][d]*x[m][e]
            }
        } else {
            #pragma unroll
            for (int c = 0; c < 5; ++c) {
                const int d = 27 + c;                       // d = 27..31
                _Float16 x0 = (_Float16)xr0[d];
                _Float16 x1 = (_Float16)xr1[d];
                do_chunk(bw, c, xe0 * x0, xe1 * x1);
            }
            do_chunk(bw, 5, xe0, xe1);                      // d==32: x vs -2m
        }
        __syncthreads();   // all waves done reading before this buffer is refilled

        if (grp == 3) {    // kgroup complete: fold 64 cluster columns into LSE
            #pragma unroll
            for (int tt = 0; tt < NTILE; ++tt) {
                const int cl = kg * 64 + tt * 16 + rlo;
                const float gvt = gvals[cl];
                #pragma unroll
                for (int mt = 0; mt < 2; ++mt) {
                    #pragma unroll
                    for (int r = 0; r < 8; ++r) {
                        float v = -0.5f * acc[tt][mt][r] + gvt;
                        if (v <= mrun[mt][r]) {
                            srun[mt][r] += __expf(v - mrun[mt][r]);
                        } else {
                            srun[mt][r] = srun[mt][r] * __expf(mrun[mt][r] - v) + 1.0f;
                            mrun[mt][r] = v;
                        }
                    }
                    acc[tt][mt] = zero;
                }
            }
        }
    }

    // merge across the 16 lanes of each half-wave (cluster columns)
    #pragma unroll
    for (int off = 1; off < 16; off <<= 1) {
        #pragma unroll
        for (int mt = 0; mt < 2; ++mt) {
            #pragma unroll
            for (int r = 0; r < 8; ++r) {
                float m2 = __shfl_xor(mrun[mt][r], off, 32);
                float s2 = __shfl_xor(srun[mt][r], off, 32);
                float M  = fmaxf(mrun[mt][r], m2);
                srun[mt][r] = srun[mt][r] * __expf(mrun[mt][r] - M)
                            + s2 * __expf(m2 - M);
                mrun[mt][r] = M;
            }
        }
    }

    if (rlo == 0) {
        const float tv = thr[0];
        #pragma unroll
        for (int mt = 0; mt < 2; ++mt) {
            const int base = row0 + mt * 16 + upper * 8;
            #pragma unroll
            for (int r = 0; r < 8; ++r)
                out[base + r] = mrun[mt][r] + __logf(srun[mt][r]) - tv;
        }
    }
}

extern "C" void kernel_launch(void* const* d_in, const int* in_sizes, int n_in,
                              void* d_out, int out_size, void* d_ws, size_t ws_size,
                              hipStream_t stream) {
    const float* points  = (const float*)d_in[0];
    const float* centers = (const float*)d_in[1];
    const float* cis     = (const float*)d_in[2];
    const float* weights = (const float*)d_in[3];
    const float* thr     = (const float*)d_in[4];
    float* out = (float*)d_out;

    float* g    = (float*)d_ws;
    char* bpack = (char*)d_ws + 1024;

    gmm_prep<<<KCL, 32, 0, stream>>>(centers, cis, weights, g, bpack);
    gmm_ll<<<NPTS / 256, 256, 0, stream>>>(points, g, bpack, thr, out);
    (void)in_sizes; (void)n_in; (void)out_size; (void)ws_size;
}
